// RNN_3058016715203
// MI455X (gfx1250) — matrix operand order repaired and checked by reading, not compiled, not run on hardware
//
#include <hip/hip_runtime.h>

#define HD 64      // hidden dim
#define G4 256     // 4*HD gates

typedef __attribute__((ext_vector_type(16))) _Float16 v16h;
typedef __attribute__((ext_vector_type(8)))  float    v8f;

__device__ __forceinline__ float fast_sigmoid(float x) {
    // safe at extremes: exp->inf gives 0, exp->0 gives 1
    return 1.0f / (1.0f + __expf(-x));
}
__device__ __forceinline__ float fast_tanh(float x) {
    // tanh(x) = sign(x) * (1 - 2/(e^{2|x|}+1)); saturates cleanly to +/-1
    float ax = fabsf(x);
    float e  = __expf(2.0f * ax);
    float r  = 1.0f - 2.0f / (e + 1.0f);
    return copysignf(r, x);
}

// ---------------------------------------------------------------------------
// Sequential LSTM recurrence: 1 block, 256 threads (8 wave32 on one WGP).
// Thread j owns gate j (i|f|g|o PyTorch order). W_hh row held in 64 VGPRs,
// W_ih row in 12 VGPRs, bias fused. h broadcast via LDS. Feature row for t+1
// prefetched into registers while computing step t.
// ---------------------------------------------------------------------------
__global__ __launch_bounds__(256) void lstm_seq_kernel(
    const float* __restrict__ feature,  // [T,12]
    const float* __restrict__ W_ih,     // [256,12]
    const float* __restrict__ W_hh,     // [256,64]
    const float* __restrict__ b_ih,     // [256]
    const float* __restrict__ b_hh,     // [256]
    const float* __restrict__ W_out,    // [2,64]
    const float* __restrict__ b_out,    // [2]
    float* __restrict__ hs,             // [T,64] workspace (may be null)
    float* __restrict__ out,            // [T,2]  (used when direct_out)
    int T, int direct_out)
{
    const int tid = threadIdx.x;

    __shared__ __align__(16) float s_h[HD];
    __shared__ float s_act[G4];

    // --- persistent per-thread weights in registers ---
    float whh[HD];
    {
        const float4* wrow = (const float4*)(W_hh + (size_t)tid * HD); // 256B aligned
        #pragma unroll
        for (int k = 0; k < HD / 4; ++k) {
            float4 w = wrow[k];
            whh[4*k+0] = w.x; whh[4*k+1] = w.y; whh[4*k+2] = w.z; whh[4*k+3] = w.w;
        }
    }
    float wih[12];
    {
        const float4* wrow = (const float4*)(W_ih + (size_t)tid * 12);  // 48B aligned
        #pragma unroll
        for (int k = 0; k < 3; ++k) {
            float4 w = wrow[k];
            wih[4*k+0] = w.x; wih[4*k+1] = w.y; wih[4*k+2] = w.z; wih[4*k+3] = w.w;
        }
    }
    const float bias = b_ih[tid] + b_hh[tid];

    // fallback-only: output weight rows for threads 0/1
    float wout[HD];
    float bo = 0.0f;
    if (direct_out && tid < 2) {
        const float4* wrow = (const float4*)(W_out + (size_t)tid * HD);
        #pragma unroll
        for (int k = 0; k < HD / 4; ++k) {
            float4 w = wrow[k];
            wout[4*k+0] = w.x; wout[4*k+1] = w.y; wout[4*k+2] = w.z; wout[4*k+3] = w.w;
        }
        bo = b_out[tid];
    }

    float c = 0.0f;
    if (tid < HD) s_h[tid] = 0.0f;
    __syncthreads();

    // feature prefetch registers (row = 48B, 16B aligned -> 3x b128)
    float4 cf0, cf1, cf2;
    {
        const float4* fr = (const float4*)feature;
        cf0 = fr[0]; cf1 = fr[1]; cf2 = fr[2];
    }

    for (int t = 0; t < T; ++t) {
        float4 nf0 = cf0, nf1 = cf1, nf2 = cf2;
        if (t + 1 < T) {
            const float4* fr = (const float4*)(feature + (size_t)(t + 1) * 12);
            nf0 = fr[0]; nf1 = fr[1]; nf2 = fr[2];   // issued early; waited at loop end
        }

        // gate_j = bias + x_t . W_ih[j] + h_{t-1} . W_hh[j]
        float g = bias;
        g = fmaf(cf0.x, wih[0], g); g = fmaf(cf0.y, wih[1], g);
        g = fmaf(cf0.z, wih[2], g); g = fmaf(cf0.w, wih[3], g);
        g = fmaf(cf1.x, wih[4], g); g = fmaf(cf1.y, wih[5], g);
        g = fmaf(cf1.z, wih[6], g); g = fmaf(cf1.w, wih[7], g);
        g = fmaf(cf2.x, wih[8], g); g = fmaf(cf2.y, wih[9], g);
        g = fmaf(cf2.z, wih[10], g); g = fmaf(cf2.w, wih[11], g);

        const float4* h4 = (const float4*)s_h;   // broadcast reads, conflict-free
        #pragma unroll
        for (int k = 0; k < HD / 4; ++k) {
            float4 hv = h4[k];
            g = fmaf(hv.x, whh[4*k+0], g);
            g = fmaf(hv.y, whh[4*k+1], g);
            g = fmaf(hv.z, whh[4*k+2], g);
            g = fmaf(hv.w, whh[4*k+3], g);
        }

        // activation: gates [128,192) are tanh (g-gate), rest sigmoid
        float a = (tid >= 2 * HD && tid < 3 * HD) ? fast_tanh(g) : fast_sigmoid(g);
        s_act[tid] = a;
        __syncthreads();

        if (tid < HD) {
            float ig = s_act[tid];
            float fg = s_act[tid + HD];
            float gg = s_act[tid + 2 * HD];
            float og = s_act[tid + 3 * HD];
            c = fmaf(fg, c, ig * gg);
            float h = og * fast_tanh(c);
            s_h[tid] = h;
            if (hs) hs[(size_t)t * HD + tid] = h;   // fire-and-forget stream
        }
        __syncthreads();

        if (direct_out && tid < 2) {
            // safe: s_h stable until all threads pass the *next* step's first barrier
            float acc = bo;
            #pragma unroll
            for (int k = 0; k < HD; ++k) acc = fmaf(s_h[k], wout[k], acc);
            out[(size_t)t * 2 + tid] = acc;
        }

        cf0 = nf0; cf1 = nf1; cf2 = nf2;
    }
}

// ---------------------------------------------------------------------------
// Output projection out = hs @ W_out^T + b_out via WMMA.
// One wave per 16-timestep tile. N padded 2->16, K=64 split into two
// v_wmma_f32_16x16x32_f16 ops. Layouts per ISA 7.12.2 (16-bit A 16x32,
// 16-bit B 32x16, 32-bit C/D 16x16). Branch-free hot path: A loaded as
// b128, B built with clamped-index + mask (cndmask, no exec divergence),
// stores guarded only on the (rare) partial last tile.
// ---------------------------------------------------------------------------
__global__ __launch_bounds__(32) void out_proj_wmma_kernel(
    const float* __restrict__ hs,     // [T,64]
    const float* __restrict__ W_out,  // [2,64]
    const float* __restrict__ b_out,  // [2]
    float* __restrict__ out,          // [T,2]
    int T)
{
    const int tile = blockIdx.x;
    const int lane = threadIdx.x;
    const int m    = lane & 15;
    const bool hi  = lane >= 16;

    int row_t = tile * 16 + m;
    if (row_t >= T) row_t = T - 1;                // clamp loads (stores guarded)
    const float4* row4 = (const float4*)(hs + (size_t)row_t * HD);  // 256B-aligned row

    // A (16x32 f16): lo lanes: elems 0..7 -> K=0..7, 8..15 -> K=16..23
    //                hi lanes: elems 0..7 -> K=8..15, 8..15 -> K=24..31
    const int ka4 = hi ? 2 : 0;          // float4 index of K-base (0 or 8 floats)
    float4 g0 = row4[ka4 + 0];           // K = ka .. ka+3
    float4 g1 = row4[ka4 + 1];           // K = ka+4 .. ka+7
    float4 g2 = row4[ka4 + 4];           // K = 16+ka ..
    float4 g3 = row4[ka4 + 5];
    float4 g4 = row4[ka4 + 8];           // K = 32+ka ..
    float4 g5 = row4[ka4 + 9];
    float4 g6 = row4[ka4 + 12];          // K = 48+ka ..
    float4 g7 = row4[ka4 + 13];

    v16h a0, a1;
    a0[0]  = (_Float16)g0.x; a0[1]  = (_Float16)g0.y; a0[2]  = (_Float16)g0.z; a0[3]  = (_Float16)g0.w;
    a0[4]  = (_Float16)g1.x; a0[5]  = (_Float16)g1.y; a0[6]  = (_Float16)g1.z; a0[7]  = (_Float16)g1.w;
    a0[8]  = (_Float16)g2.x; a0[9]  = (_Float16)g2.y; a0[10] = (_Float16)g2.z; a0[11] = (_Float16)g2.w;
    a0[12] = (_Float16)g3.x; a0[13] = (_Float16)g3.y; a0[14] = (_Float16)g3.z; a0[15] = (_Float16)g3.w;
    a1[0]  = (_Float16)g4.x; a1[1]  = (_Float16)g4.y; a1[2]  = (_Float16)g4.z; a1[3]  = (_Float16)g4.w;
    a1[4]  = (_Float16)g5.x; a1[5]  = (_Float16)g5.y; a1[6]  = (_Float16)g5.z; a1[7]  = (_Float16)g5.w;
    a1[8]  = (_Float16)g6.x; a1[9]  = (_Float16)g6.y; a1[10] = (_Float16)g6.z; a1[11] = (_Float16)g6.w;
    a1[12] = (_Float16)g7.x; a1[13] = (_Float16)g7.y; a1[14] = (_Float16)g7.z; a1[15] = (_Float16)g7.w;

    // B (32x16 f16): N = lane&15; lo lanes: elems 0..7 -> K=0..7, 8..15 -> K=16..23; hi: K=8..15, K=24..31.
    // Branch-free: clamp row index, zero out padded columns by multiply.
    const int  n     = m;
    const int  nc    = (n < 2) ? n : 0;
    const float nmask = (n < 2) ? 1.0f : 0.0f;
    const int  kb4   = hi ? 2 : 0;       // float4 index of K-base (0 or 8 floats)
    const float4* wrow4 = (const float4*)(W_out + (size_t)nc * HD);

    float4 w0 = wrow4[kb4 + 0];          // K = kb .. kb+3
    float4 w1 = wrow4[kb4 + 1];
    float4 w2 = wrow4[kb4 + 4];
    float4 w3 = wrow4[kb4 + 5];
    float4 w4 = wrow4[kb4 + 8];          // K = kb+32 ..
    float4 w5 = wrow4[kb4 + 9];
    float4 w6 = wrow4[kb4 + 12];
    float4 w7 = wrow4[kb4 + 13];

    v16h b0, b1;
    b0[0]  = (_Float16)(w0.x * nmask); b0[1]  = (_Float16)(w0.y * nmask);
    b0[2]  = (_Float16)(w0.z * nmask); b0[3]  = (_Float16)(w0.w * nmask);
    b0[4]  = (_Float16)(w1.x * nmask); b0[5]  = (_Float16)(w1.y * nmask);
    b0[6]  = (_Float16)(w1.z * nmask); b0[7]  = (_Float16)(w1.w * nmask);
    b0[8]  = (_Float16)(w2.x * nmask); b0[9]  = (_Float16)(w2.y * nmask);
    b0[10] = (_Float16)(w2.z * nmask); b0[11] = (_Float16)(w2.w * nmask);
    b0[12] = (_Float16)(w3.x * nmask); b0[13] = (_Float16)(w3.y * nmask);
    b0[14] = (_Float16)(w3.z * nmask); b0[15] = (_Float16)(w3.w * nmask);
    b1[0]  = (_Float16)(w4.x * nmask); b1[1]  = (_Float16)(w4.y * nmask);
    b1[2]  = (_Float16)(w4.z * nmask); b1[3]  = (_Float16)(w4.w * nmask);
    b1[4]  = (_Float16)(w5.x * nmask); b1[5]  = (_Float16)(w5.y * nmask);
    b1[6]  = (_Float16)(w5.z * nmask); b1[7]  = (_Float16)(w5.w * nmask);
    b1[8]  = (_Float16)(w6.x * nmask); b1[9]  = (_Float16)(w6.y * nmask);
    b1[10] = (_Float16)(w6.z * nmask); b1[11] = (_Float16)(w6.w * nmask);
    b1[12] = (_Float16)(w7.x * nmask); b1[13] = (_Float16)(w7.y * nmask);
    b1[14] = (_Float16)(w7.z * nmask); b1[15] = (_Float16)(w7.w * nmask);

    v8f acc = {};
    acc = __builtin_amdgcn_wmma_f32_16x16x32_f16(false, a0, false, b0,
                                                 (short)0, acc, false, false);
    acc = __builtin_amdgcn_wmma_f32_16x16x32_f16(false, a1, false, b1,
                                                 (short)0, acc, false, false);

    // D (16x16 f32): lane<16 -> N=lane, M=v; lane>=16 -> N=lane-16, M=v+8
    const bool full = (tile * 16 + 15) < T;       // uniform: only last tile partial
    if (n < 2) {
        const int mbase = hi ? 8 : 0;
        const float bb = b_out[n];
        if (full) {
            #pragma unroll
            for (int v = 0; v < 8; ++v)
                out[(size_t)(tile * 16 + mbase + v) * 2 + n] = acc[v] + bb;
        } else {
            #pragma unroll
            for (int v = 0; v < 8; ++v) {
                int tt = tile * 16 + mbase + v;
                if (tt < T) out[(size_t)tt * 2 + n] = acc[v] + bb;
            }
        }
    }
}

extern "C" void kernel_launch(void* const* d_in, const int* in_sizes, int n_in,
                              void* d_out, int out_size, void* d_ws, size_t ws_size,
                              hipStream_t stream) {
    const float* feature = (const float*)d_in[0];
    const float* W_ih    = (const float*)d_in[1];
    const float* W_hh    = (const float*)d_in[2];
    const float* b_ih    = (const float*)d_in[3];
    const float* b_hh    = (const float*)d_in[4];
    const float* W_out   = (const float*)d_in[5];
    const float* b_out   = (const float*)d_in[6];
    float* out = (float*)d_out;

    const int T = in_sizes[0] / 12;
    const size_t hs_bytes = (size_t)T * HD * sizeof(float);
    const bool use_ws = (d_ws != nullptr) && (ws_size >= hs_bytes);
    float* hs = use_ws ? (float*)d_ws : nullptr;

    lstm_seq_kernel<<<1, 256, 0, stream>>>(
        feature, W_ih, W_hh, b_ih, b_hh, W_out, b_out,
        hs, out, T, use_ws ? 0 : 1);

    if (use_ws) {
        const int tiles = (T + 15) / 16;
        out_proj_wmma_kernel<<<tiles, 32, 0, stream>>>(hs, W_out, b_out, out, T);
    }
}
